// ReyEquiv2to2_65790309040638
// MI455X (gfx1250) — matrix operand order repaired and checked by reading, not compiled, not run on hardware
//
#include <hip/hip_runtime.h>
#include <hip/hip_bf16.h>

typedef __attribute__((ext_vector_type(16))) __bf16 bf16x16;
typedef __attribute__((ext_vector_type(8)))  __bf16 bf16x8;
typedef __attribute__((ext_vector_type(8)))  float  f32x8;

// ---------- bf16 helpers: use native hardware converts (RNE) ----------
__device__ __forceinline__ __bf16 f2bf(float f) { return (__bf16)f; }
__device__ __forceinline__ float  bf2f(__bf16 h) { return (float)h; }
__device__ __forceinline__ unsigned short f2bfbits(float f) {
  __bf16 h = (__bf16)f;
  return __builtin_bit_cast(unsigned short, h);
}

// =====================================================================
// Generic WMMA MLP layer: out[rows=128][NCOLS] = act(in[128][KDIM] @ W + b)
// A in LDS row-major (stride aStride), W in global pre-transposed bf16
// Wt[col][k] (k contiguous), out to LDS row-major (stride oStride).
// 8 waves, wave w owns rows w*16..w*16+15. NCOLS processed in chunks of
// <=8 16-wide col tiles (8 f32x8 accumulators). Bias is pre-loaded into
// the accumulators (each lane owns one output column across its 8 regs).
// =====================================================================
template<int KDIM, int NCOLS, bool RELU>
__device__ __forceinline__ void wmma_mlp_layer(
    const __bf16* aIn, int aStride,
    const __bf16* __restrict__ Wt,
    const float*  __restrict__ bias,
    __bf16* aOut, int oStride,
    int wave, int lane)
{
  const int rowbase = wave * 16;
  const int m  = lane & 15;
  const int hi = (lane & 16) ? 1 : 0;
  const int klo = hi ? 16 : 0;   // A-frag K offset within 32-K step
  const int kb  = hi ? 16 : 0;   // B-frag K offset within 32-K step
  constexpr int NT_TOT = NCOLS / 16;
  constexpr int NT = NT_TOT < 8 ? NT_TOT : 8;

  for (int nt0 = 0; nt0 < NT_TOT; nt0 += NT) {
    f32x8 acc[NT];
#pragma unroll
    for (int q = 0; q < NT; ++q) {
      const float bv = bias[(nt0 + q) * 16 + m];
      acc[q] = (f32x8){bv, bv, bv, bv, bv, bv, bv, bv};
    }

#pragma unroll
    for (int ks = 0; ks < KDIM / 32; ++ks) {
      const __bf16* arow = aIn + (rowbase + m) * aStride + ks * 32;
      bf16x8 alo = *(const bf16x8*)(arow + klo);
      bf16x8 ahi = *(const bf16x8*)(arow + klo + 8);
      bf16x16 afrag;
#pragma unroll
      for (int e = 0; e < 8; ++e) { afrag[e] = alo[e]; afrag[e + 8] = ahi[e]; }
#pragma unroll
      for (int q = 0; q < NT; ++q) {
        const int col = (nt0 + q) * 16 + m;
        bf16x16 bfrag = *(const bf16x16*)(Wt + col * KDIM + ks * 32 + kb);
        acc[q] = __builtin_amdgcn_wmma_f32_16x16x32_bf16(
            false, afrag, false, bfrag, (short)0, acc[q], false, false);
      }
    }
    // epilogue: relu + convert + store (D layout: lane holds col m,
    // VGPR r holds row r (lanes 0-15) or r+8 (lanes 16-31))
#pragma unroll
    for (int q = 0; q < NT; ++q) {
      const int col = (nt0 + q) * 16 + m;
#pragma unroll
      for (int r = 0; r < 8; ++r) {
        const int row = rowbase + (hi ? r + 8 : r);
        float v = acc[q][r];
        if (RELU) v = fmaxf(v, 0.0f);
        aOut[row * oStride + col] = f2bf(v);
      }
    }
  }
}

// =====================================================================
// Kernel 0: zero mid buffer, convert+transpose weights to bf16 Wt[out][in]
// =====================================================================
__global__ __launch_bounds__(256) void prep_kernel(
    const float* __restrict__ W2, const float* __restrict__ W3,
    const float* __restrict__ Wc1, const float* __restrict__ Wc2,
    const float* __restrict__ Wc3, const float* __restrict__ Wc4,
    float* __restrict__ mid,
    unsigned short* __restrict__ W2t,  unsigned short* __restrict__ W3t,
    unsigned short* __restrict__ Wc1t, unsigned short* __restrict__ Wc2t,
    unsigned short* __restrict__ Wc3t, unsigned short* __restrict__ Wc4t)
{
  const unsigned idx = blockIdx.x * 256u + threadIdx.x;
  if (idx < 4u * 64u * 4096u) mid[idx] = 0.0f;            // 1,048,576 f32
  if (idx < 16384u) {                                     // W2,W3: 128x128
    unsigned o = idx >> 7, i = idx & 127u;
    W2t[idx] = f2bfbits(W2[i * 128u + o]);
    W3t[idx] = f2bfbits(W3[i * 128u + o]);
  }
  if (idx < 8192u) {                                      // Wc1: 64x128 -> [128][64]
    unsigned o = idx / 64u, i = idx & 63u;
    Wc1t[idx] = f2bfbits(Wc1[i * 128u + o]);
  }
  if (idx < 32768u) {                                     // Wc2: 128x256 -> [256][128]
    unsigned o = idx >> 7, i = idx & 127u;
    Wc2t[idx] = f2bfbits(Wc2[i * 256u + o]);
  }
  if (idx < 65536u) {                                     // Wc3: 256x256 -> [256][256]
    unsigned o = idx >> 8, i = idx & 255u;
    Wc3t[idx] = f2bfbits(Wc3[i * 256u + o]);
  }
  if (idx < 8192u) {                                      // Wc4: 256x32 -> [32][256]
    unsigned o = idx >> 8, i = idx & 255u;
    Wc4t[idx] = f2bfbits(Wc4[i * 32u + o]);
  }
}

// =====================================================================
// Kernel 1: per-element MLP (4->128->128->128->2) + scatter into mid.
// grid = (256 bc-pairs, 32 tiles of 128 rows), block = 256 threads.
// =====================================================================
__global__ __launch_bounds__(256) void elem_mlp_kernel(
    const float* __restrict__ x,
    const float* __restrict__ W1, const float* __restrict__ b1,
    const unsigned short* __restrict__ W2tb, const float* __restrict__ b2,
    const unsigned short* __restrict__ W3tb, const float* __restrict__ b3,
    const float* __restrict__ W4, const float* __restrict__ b4,
    float* __restrict__ mid)
{
  __shared__ float  sFeat[128][4];
  __shared__ __bf16 act0[128 * 128];
  __shared__ __bf16 act1[128 * 128];

  const int bc   = blockIdx.x;        // 0..255 (b*64+c)
  const int tile = blockIdx.y;        // 0..31
  const int rbase = tile * 128;
  const int t = threadIdx.x, lane = t & 31, wave = t >> 5;

  // ---- phase 1a: gather the 4 features per row ----
  if (t < 128) {
    const int rl = rbase + t;
    const bool valid = rl < 4032;     // 63*64 rows per (b,c)
    const int rr = valid ? rl : 0;
    const int ii = rr >> 6, kk = rr & 63;
    const int jj = (kk + ii + 1) & 63;
    const float* xb = x + bc * 4096;
    float f0 = xb[kk * 64 + kk];
    float f1 = xb[kk * 64 + jj];
    float f2 = xb[jj * 64 + kk];
    float f3 = xb[jj * 64 + jj];
    if (!valid) { f0 = f1 = f2 = f3 = 0.f; }
    sFeat[t][0] = f0; sFeat[t][1] = f1; sFeat[t][2] = f2; sFeat[t][3] = f3;
  }
  __syncthreads();

  // ---- phase 1b: layer1 (K=4) in VALU f32, write bf16 act0 ----
  {
    const int row = t >> 1;
    const int c0  = (t & 1) * 64;
    const float f0 = sFeat[row][0], f1 = sFeat[row][1];
    const float f2 = sFeat[row][2], f3 = sFeat[row][3];
    for (int c = c0; c < c0 + 64; ++c) {
      float a = b1[c];
      a = fmaf(f0, W1[c], a);
      a = fmaf(f1, W1[128 + c], a);
      a = fmaf(f2, W1[256 + c], a);
      a = fmaf(f3, W1[384 + c], a);
      act0[row * 128 + c] = f2bf(fmaxf(a, 0.f));
    }
  }
  __syncthreads();

  // ---- layers 2 & 3: 128x128x128 WMMA GEMMs ----
  wmma_mlp_layer<128, 128, true>(act0, 128, (const __bf16*)W2tb, b2, act1, 128, wave, lane);
  __syncthreads();
  wmma_mlp_layer<128, 128, true>(act1, 128, (const __bf16*)W3tb, b3, act0, 128, wave, lane);
  __syncthreads();

  // ---- layer 4 (N=2) in VALU + scatter ----
  if (t < 128) {
    const int rl = rbase + t;
    if (rl < 4032) {
      float a0 = b4[0], a1 = b4[1];
#pragma unroll
      for (int kc = 0; kc < 16; ++kc) {
        bf16x8 v = *(const bf16x8*)(act0 + t * 128 + kc * 8);
#pragma unroll
        for (int e = 0; e < 8; ++e) {
          const float vv = bf2f(v[e]);
          const int k = kc * 8 + e;
          a0 = fmaf(vv, W4[2 * k + 0], a0);
          a1 = fmaf(vv, W4[2 * k + 1], a1);
        }
      }
      const int ii = rl >> 6, kk = rl & 63;
      const int jj = (kk + ii + 1) & 63;
      float* mb = mid + bc * 4096;
      mb[kk * 64 + jj] = a1;                                    // off-diagonal
      unsafeAtomicAdd(&mb[kk * 64 + kk], a0 * (1.0f / 63.0f));  // diagonal mean
    }
  }
}

// =====================================================================
// Kernel 2: per-pixel channel MLP 64->128->256->256->32 (WMMA).
// grid = (4 batches, 32 tiles of 128 pixels), block = 256 threads.
// =====================================================================
__global__ __launch_bounds__(256) void chan_mlp_kernel(
    const float* __restrict__ mid,
    const unsigned short* __restrict__ Wc1t, const float* __restrict__ bc1,
    const unsigned short* __restrict__ Wc2t, const float* __restrict__ bc2,
    const unsigned short* __restrict__ Wc3t, const float* __restrict__ bc3,
    const unsigned short* __restrict__ Wc4t, const float* __restrict__ bc4,
    float* __restrict__ out)
{
  __shared__ __bf16 bufA[128 * 256];
  __shared__ __bf16 bufB[128 * 256];

  const int b  = blockIdx.x;          // 0..3
  const int pt = blockIdx.y;          // 0..31
  const int pixbase = pt * 128;
  const int t = threadIdx.x, lane = t & 31, wave = t >> 5;

  // stage input transposed: bufA[pixel][channel] (stride 256)
  for (int idx = t; idx < 64 * 128; idx += 256) {
    const int c = idx >> 7, p = idx & 127;
    bufA[p * 256 + c] = f2bf(mid[(b * 64 + c) * 4096 + pixbase + p]);
  }
  __syncthreads();

  wmma_mlp_layer< 64, 128, true>(bufA, 256, (const __bf16*)Wc1t, bc1, bufB, 256, wave, lane);
  __syncthreads();
  wmma_mlp_layer<128, 256, true>(bufB, 256, (const __bf16*)Wc2t, bc2, bufA, 256, wave, lane);
  __syncthreads();
  wmma_mlp_layer<256, 256, true>(bufA, 256, (const __bf16*)Wc3t, bc3, bufB, 256, wave, lane);
  __syncthreads();

  // final layer: K=256, cols=32, f32 store straight to out[b][d][pixel]
  {
    const __bf16* Wt = (const __bf16*)Wc4t;
    const int rowbase = wave * 16;
    const int m  = lane & 15;
    const int hi = (lane & 16) ? 1 : 0;
    const int klo = hi ? 16 : 0, kb = hi ? 16 : 0;
    f32x8 acc[2];
#pragma unroll
    for (int q = 0; q < 2; ++q) {
      const float bv = bc4[q * 16 + m];
      acc[q] = (f32x8){bv, bv, bv, bv, bv, bv, bv, bv};
    }
#pragma unroll
    for (int ks = 0; ks < 8; ++ks) {
      const __bf16* arow = bufB + (rowbase + m) * 256 + ks * 32;
      bf16x8 alo = *(const bf16x8*)(arow + klo);
      bf16x8 ahi = *(const bf16x8*)(arow + klo + 8);
      bf16x16 afrag;
#pragma unroll
      for (int e = 0; e < 8; ++e) { afrag[e] = alo[e]; afrag[e + 8] = ahi[e]; }
#pragma unroll
      for (int q = 0; q < 2; ++q) {
        bf16x16 bfrag = *(const bf16x16*)(Wt + (q * 16 + m) * 256 + ks * 32 + kb);
        acc[q] = __builtin_amdgcn_wmma_f32_16x16x32_bf16(
            false, afrag, false, bfrag, (short)0, acc[q], false, false);
      }
    }
#pragma unroll
    for (int q = 0; q < 2; ++q) {
      const int dch = q * 16 + m;
#pragma unroll
      for (int r = 0; r < 8; ++r) {
        const int row = rowbase + (hi ? r + 8 : r);
        out[(b * 32 + dch) * 4096 + pixbase + row] = acc[q][r];
      }
    }
  }
}

// =====================================================================
extern "C" void kernel_launch(void* const* d_in, const int* in_sizes, int n_in,
                              void* d_out, int out_size, void* d_ws, size_t ws_size,
                              hipStream_t stream) {
  (void)in_sizes; (void)n_in; (void)out_size; (void)ws_size;
  const float* x   = (const float*)d_in[0];
  const float* W1  = (const float*)d_in[1];
  const float* b1  = (const float*)d_in[2];
  const float* W2  = (const float*)d_in[3];
  const float* b2  = (const float*)d_in[4];
  const float* W3  = (const float*)d_in[5];
  const float* b3  = (const float*)d_in[6];
  const float* W4  = (const float*)d_in[7];
  const float* b4  = (const float*)d_in[8];
  const float* Wc1 = (const float*)d_in[9];
  const float* bc1 = (const float*)d_in[10];
  const float* Wc2 = (const float*)d_in[11];
  const float* bc2 = (const float*)d_in[12];
  const float* Wc3 = (const float*)d_in[13];
  const float* bc3 = (const float*)d_in[14];
  const float* Wc4 = (const float*)d_in[15];
  const float* bc4 = (const float*)d_in[16];

  char* ws = (char*)d_ws;
  float*          mid  = (float*)(ws);                         // 4,194,304 B
  unsigned short* W2t  = (unsigned short*)(ws + 4194304);      //    32,768 B
  unsigned short* W3t  = (unsigned short*)(ws + 4227072);      //    32,768 B
  unsigned short* Wc1t = (unsigned short*)(ws + 4259840);      //    16,384 B
  unsigned short* Wc2t = (unsigned short*)(ws + 4276224);      //    65,536 B
  unsigned short* Wc3t = (unsigned short*)(ws + 4341760);      //   131,072 B
  unsigned short* Wc4t = (unsigned short*)(ws + 4472832);      //    16,384 B

  prep_kernel<<<4096, 256, 0, stream>>>(W2, W3, Wc1, Wc2, Wc3, Wc4,
                                        mid, W2t, W3t, Wc1t, Wc2t, Wc3t, Wc4t);
  elem_mlp_kernel<<<dim3(256, 32), 256, 0, stream>>>(
      x, W1, b1, W2t, b2, W3t, b3, W4, b4, mid);
  chan_mlp_kernel<<<dim3(4, 32), 256, 0, stream>>>(
      mid, Wc1t, bc1, Wc2t, bc2, Wc3t, bc3, Wc4t, bc4, (float*)d_out);
}
